// Attention_54065048322573
// MI455X (gfx1250) — matrix operand order repaired and checked by reading, not compiled, not run on hardware
//
#include <hip/hip_runtime.h>
#include <hip/hip_bf16.h>
#include <stdint.h>

// ---------------------------------------------------------------------------
// Problem constants (from reference)
// ---------------------------------------------------------------------------
#define BATCH 4
#define CDIM 256
#define HEADS 8
#define CH 32            // channels per head
#define IMGH 160
#define IMGW 160
#define HW (IMGH * IMGW) // 25600
#define EPS 1e-12f

typedef unsigned short u16;
typedef __attribute__((ext_vector_type(16))) __bf16 v16bf;
typedef __attribute__((ext_vector_type(8)))  float  v8f;
typedef __attribute__((ext_vector_type(4)))  unsigned int v4u;

union Frag { v16bf v; v4u u[2]; };

static __device__ inline float bf2f(u16 a) {
    union { unsigned int i; float f; } z; z.i = ((unsigned int)a) << 16; return z.f;
}
static __device__ inline u16 f2bf(float f) {
    union { float f; unsigned int i; } z; z.f = f;
    unsigned int r = z.i + 0x7FFFu + ((z.i >> 16) & 1u);   // round to nearest even
    return (u16)(r >> 16);
}
static __device__ inline v8f wmma_bf16(const Frag& a, const Frag& b, v8f c) {
    // D = A(16x32 bf16) * B(32x16 bf16) + C(16x16 f32)
    return __builtin_amdgcn_wmma_f32_16x16x32_bf16(false, a.v, false, b.v,
                                                   (short)0, c, false, false);
}
static __device__ inline v8f zero_v8f() {
    v8f z = {0.f,0.f,0.f,0.f,0.f,0.f,0.f,0.f}; return z;
}

// ---------------------------------------------------------------------------
// Tiled WMMA GEMM:  Y[b] = W[MxK] * X[b][KxN]
//   block = 256 threads (8 waves), tile 128(M) x 128(N), K-step 32
//   wave (wm in 0..3, wn in 0..1) computes 32(M) x 64(N) = 2x4 WMMA tiles
//   IN_F32:  X is f32 (converted to bf16 on LDS stage) else bf16
//   OUT_F32: Y is f32 else bf16
// ---------------------------------------------------------------------------
#define LPAD 40  // LDS row pitch (elements): 80B -> 16B aligned, bank-spread

template <bool IN_F32, bool OUT_F32>
__global__ __launch_bounds__(256) void gemm_wmma_k(
    const float* __restrict__ W,   // [M][K] f32 weights
    const void*  __restrict__ Xv,  // [B][K][N]
    void*        __restrict__ Yv,  // [B][M][N]
    int M, int N, int K)
{
    __shared__ u16 sA[128 * LPAD]; // [m][k], row pitch LPAD
    __shared__ u16 sB[128 * LPAD]; // [n][k] (transposed), row pitch LPAD

    const int tid = threadIdx.x;
    const int n0 = blockIdx.x * 128;
    const int m0 = blockIdx.y * 128;
    const int b  = blockIdx.z;

    const float* Xf = (const float*)Xv + (size_t)b * K * N;
    const u16*   Xb = (const u16*)Xv   + (size_t)b * K * N;
    float* Yf = (float*)Yv + (size_t)b * M * N;
    u16*   Yb = (u16*)Yv   + (size_t)b * M * N;

    const int wave = tid >> 5, lane = tid & 31;
    const int wm = wave & 3, wn = wave >> 2;
    const int g = lane >> 4, h16 = lane & 15;

    v8f acc[2][4];
#pragma unroll
    for (int i = 0; i < 2; ++i)
#pragma unroll
        for (int j = 0; j < 4; ++j) acc[i][j] = zero_v8f();

    for (int k0 = 0; k0 < K; k0 += 32) {
        // stage A tile (128x32) from weights, cvt f32->bf16
#pragma unroll
        for (int i = 0; i < 16; ++i) {
            int idx = tid + i * 256;
            int mm = idx >> 5, kk = idx & 31;
            sA[mm * LPAD + kk] = f2bf(W[(size_t)(m0 + mm) * K + k0 + kk]);
        }
        // stage B tile (32x128) transposed into LDS [n][k]
#pragma unroll
        for (int i = 0; i < 16; ++i) {
            int idx = tid + i * 256;
            int kk = idx >> 7, nn = idx & 127;
            size_t go = (size_t)(k0 + kk) * N + n0 + nn;
            float x = IN_F32 ? Xf[go] : bf2f(Xb[go]);
            sB[nn * LPAD + kk] = f2bf(x);
            if (k0 + 32 < K) { // prefetch next k-step (global_prefetch_b8)
                if (IN_F32) __builtin_prefetch(&Xf[go + (size_t)32 * N], 0, 1);
                else        __builtin_prefetch(&Xb[go + (size_t)32 * N], 0, 1);
            }
        }
        __syncthreads();

        Frag afr[2], bfr[4];
#pragma unroll
        for (int mt = 0; mt < 2; ++mt) {
            int mr = wm * 32 + mt * 16 + h16;
            afr[mt].u[0] = *(const v4u*)&sA[mr * LPAD + g * 8];
            afr[mt].u[1] = *(const v4u*)&sA[mr * LPAD + 16 + g * 8];
        }
#pragma unroll
        for (int nt = 0; nt < 4; ++nt) {
            int nc = wn * 64 + nt * 16 + h16;
            bfr[nt].u[0] = *(const v4u*)&sB[nc * LPAD + g * 8];
            bfr[nt].u[1] = *(const v4u*)&sB[nc * LPAD + 16 + g * 8];
        }
#pragma unroll
        for (int mt = 0; mt < 2; ++mt)
#pragma unroll
            for (int nt = 0; nt < 4; ++nt)
                acc[mt][nt] = wmma_bf16(afr[mt], bfr[nt], acc[mt][nt]);
        __syncthreads();
    }

    // epilogue: D layout -> row = r + 8*g, col = lane%16 within 16x16 tile
#pragma unroll
    for (int mt = 0; mt < 2; ++mt)
#pragma unroll
        for (int nt = 0; nt < 4; ++nt)
#pragma unroll
            for (int r = 0; r < 8; ++r) {
                int row = m0 + wm * 32 + mt * 16 + g * 8 + r;
                int col = n0 + wn * 64 + nt * 16 + h16;
                float val = acc[mt][nt][r];
                if (OUT_F32) Yf[(size_t)row * N + col] = val;
                else         Yb[(size_t)row * N + col] = f2bf(val);
            }
}

// ---------------------------------------------------------------------------
// Depthwise 3x3 conv, zero padding. bf16 in/out, f32 math.
// grid: (HW/256, C, B)
// ---------------------------------------------------------------------------
__global__ __launch_bounds__(256) void dwconv3x3_k(
    const u16* __restrict__ X, const float* __restrict__ Wdw,
    u16* __restrict__ Y, int C)
{
    int p = blockIdx.x * 256 + threadIdx.x;
    int c = blockIdx.y, b = blockIdx.z;
    int y = p / IMGW, x = p % IMGW;
    size_t base = ((size_t)b * C + c) * HW;
    const float* w = Wdw + c * 9;
    float s = 0.f;
#pragma unroll
    for (int dy = -1; dy <= 1; ++dy)
#pragma unroll
        for (int dx = -1; dx <= 1; ++dx) {
            int yy = y + dy, xx = x + dx;
            if (yy >= 0 && yy < IMGH && xx >= 0 && xx < IMGW)
                s += bf2f(X[base + yy * IMGW + xx]) * w[(dy + 1) * 3 + (dx + 1)];
        }
    Y[base + p] = f2bf(s);
}

// ---------------------------------------------------------------------------
// Row L2 norms: rq[b][c] = 1/max(||q[b,c,:]||, eps); same for k (kv ch 0..255)
// grid: 2*B*C blocks of 256
// ---------------------------------------------------------------------------
__global__ __launch_bounds__(256) void rownorm_k(
    const u16* __restrict__ Q, const u16* __restrict__ KV,
    float* __restrict__ RQ, float* __restrict__ RK)
{
    __shared__ float red[256];
    int r = blockIdx.x;
    int sel = r >> 10;            // 0 = q, 1 = k
    int idx = r & 1023;           // b*256 + c
    int b = idx >> 8, c = idx & 255;
    const u16* src = (sel == 0) ? (Q + (size_t)idx * HW)
                                : (KV + ((size_t)b * (2 * CDIM) + c) * HW);
    float s = 0.f;
#pragma unroll 4
    for (int i = 0; i < HW / 256; ++i) {
        float v = bf2f(src[threadIdx.x + i * 256]);
        s += v * v;
    }
    red[threadIdx.x] = s;
    __syncthreads();
    for (int off = 128; off > 0; off >>= 1) {
        if ((int)threadIdx.x < off) red[threadIdx.x] += red[threadIdx.x + off];
        __syncthreads();
    }
    if (threadIdx.x == 0) {
        float inv = 1.0f / fmaxf(sqrtf(red[0]), EPS);
        if (sel == 0) RQ[idx] = inv; else RK[idx] = inv;
    }
}

// ---------------------------------------------------------------------------
// S[b,h] = q[b,h] * k[b,h]^T  (raw, unnormalized)  — split-K WMMA + f32 atomics
// grid: (25 n-chunks of 1024, 32 bh) x 256 threads
// wave: tile-pair (mt,nt) = wave&3, k-subchunk = wave>>2 (512 n each)
// ---------------------------------------------------------------------------
__global__ __launch_bounds__(256) void attn_qk_k(
    const u16* __restrict__ Q, const u16* __restrict__ KV,
    float* __restrict__ S)
{
    const int bh = blockIdx.y;
    const int b = bh >> 3, h = bh & 7;
    const int wave = threadIdx.x >> 5, lane = threadIdx.x & 31;
    const int mt = (wave & 3) >> 1, nt = wave & 1, ks = wave >> 2;
    const int g = lane >> 4, h16 = lane & 15;

    const int nbase = blockIdx.x * 1024 + ks * 512;

    // A: q row (c = mt*16 + h16), K-offsets g*8 and 16+g*8 (16B each)
    const u16* qrow = Q + ((size_t)b * CDIM + h * CH + mt * 16 + h16) * HW;
    // B: k row (d = h16), K-offsets g*8 and 16+g*8 (16B each)
    const u16* krow = KV + ((size_t)b * (2 * CDIM) + h * CH + h16) * HW;

    v8f acc = zero_v8f();
#pragma unroll 4
    for (int n = nbase; n < nbase + 512; n += 32) {
        Frag a, bb;
        a.u[0]  = *(const v4u*)(qrow + n + g * 8);
        a.u[1]  = *(const v4u*)(qrow + n + 16 + g * 8);
        bb.u[0] = *(const v4u*)(krow + n + g * 8);
        bb.u[1] = *(const v4u*)(krow + n + 16 + g * 8);
        acc = wmma_bf16(a, bb, acc);
    }
#pragma unroll
    for (int r = 0; r < 8; ++r) {
        int row = mt * 16 + g * 8 + r;
        int col = nt * 16 + h16;
        atomicAdd(&S[((size_t)bh * CH + row) * CH + col], acc[r]);
    }
}

// ---------------------------------------------------------------------------
// softmax row kernel: fold in 1/||q||, 1/||k||, temperature; emit bf16 probs
// one wave (32 lanes) per (b,h,c) row; grid 128 x 256
// ---------------------------------------------------------------------------
__global__ __launch_bounds__(256) void softmax_k(
    const float* __restrict__ S, const float* __restrict__ RQ,
    const float* __restrict__ RK, const float* __restrict__ temp,
    u16* __restrict__ P)
{
    int row = blockIdx.x * 8 + (threadIdx.x >> 5); // 0..1023
    int lane = threadIdx.x & 31;                   // d
    int bh = row >> 5, c = row & 31;
    int b = bh >> 3, h = bh & 7;

    float s = S[((size_t)bh * CH + c) * CH + lane]
            * RQ[b * CDIM + h * CH + c]
            * RK[b * CDIM + h * CH + lane]
            * temp[h];
    float m = s;
    for (int off = 16; off > 0; off >>= 1) m = fmaxf(m, __shfl_xor(m, off, 32));
    float e = __expf(s - m);
    float sum = e;
    for (int off = 16; off > 0; off >>= 1) sum += __shfl_xor(sum, off, 32);
    P[((size_t)bh * CH + c) * CH + lane] = f2bf(e / sum);
}

// ---------------------------------------------------------------------------
// out[b,h] = P(32x32) * v(32xHW)  — one WMMA (K=32) per 16x16 output tile
// grid: (HW/128, 32 bh) x 256; wave w owns n-tile w; v staged transposed in LDS
// ---------------------------------------------------------------------------
__global__ __launch_bounds__(256) void attn_v_k(
    const u16* __restrict__ P, const u16* __restrict__ KV,
    u16* __restrict__ O)
{
    __shared__ u16 sV[128 * LPAD]; // [n][d]
    const int bh = blockIdx.y;
    const int b = bh >> 3, h = bh & 7;
    const int n0 = blockIdx.x * 128;
    const int tid = threadIdx.x;
    const int wave = tid >> 5, lane = tid & 31;
    const int g = lane >> 4, h16 = lane & 15;

    const u16* vbase = KV + ((size_t)b * (2 * CDIM) + CDIM + h * CH) * HW;
#pragma unroll
    for (int i = 0; i < 16; ++i) {
        int idx = tid + i * 256;
        int d = idx >> 7, nn = idx & 127;
        sV[nn * LPAD + d] = vbase[(size_t)d * HW + n0 + nn];
    }
    __syncthreads();

    Frag bb;
    bb.u[0] = *(const v4u*)&sV[(wave * 16 + h16) * LPAD + g * 8];
    bb.u[1] = *(const v4u*)&sV[(wave * 16 + h16) * LPAD + 16 + g * 8];

#pragma unroll
    for (int mt = 0; mt < 2; ++mt) {
        const u16* prow = P + ((size_t)bh * CH + mt * 16 + h16) * CH;
        Frag a;
        a.u[0] = *(const v4u*)(prow + g * 8);
        a.u[1] = *(const v4u*)(prow + 16 + g * 8);
        v8f acc = wmma_bf16(a, bb, zero_v8f());
#pragma unroll
        for (int r = 0; r < 8; ++r) {
            int row = h * CH + mt * 16 + g * 8 + r;
            int col = n0 + wave * 16 + h16;
            O[((size_t)b * CDIM + row) * HW + col] = f2bf(acc[r]);
        }
    }
}

// ---------------------------------------------------------------------------
// Host launcher
// ---------------------------------------------------------------------------
extern "C" void kernel_launch(void* const* d_in, const int* in_sizes, int n_in,
                              void* d_out, int out_size, void* d_ws, size_t ws_size,
                              hipStream_t stream) {
    const float* low    = (const float*)d_in[0];
    const float* high   = (const float*)d_in[1];
    const float* q_w    = (const float*)d_in[2];
    const float* q_dw_w = (const float*)d_in[3];
    const float* kv_w   = (const float*)d_in[4];
    const float* kv_dw_w= (const float*)d_in[5];
    const float* proj_w = (const float*)d_in[6];
    const float* temp   = (const float*)d_in[7];
    float* out = (float*)d_out;

    char* ws = (char*)d_ws;
    const size_t szC  = (size_t)BATCH * CDIM * HW * 2;     // 52.4 MB (bf16)
    const size_t sz2C = 2 * szC;                           // 104.8 MB
    u16* Xq   = (u16*)(ws);                                // conv1x1(high); later reused as attn output
    u16* Xkv  = (u16*)(ws + szC);                          // conv1x1(low)
    u16* Qb   = (u16*)(ws + szC + sz2C);                   // q after dwconv
    u16* KVb  = (u16*)(ws + 2 * szC + sz2C);               // kv after dwconv (k ch 0..255, v 256..511)
    char* tail = ws + 2 * szC + 2 * sz2C;
    float* RQ = (float*)(tail);
    float* RK = (float*)(tail + 4096);
    float* S  = (float*)(tail + 8192);                     // [32bh][32][32] f32
    u16*   P  = (u16*)(tail + 8192 + 131072);              // bf16 probs
    u16* OA   = Xq;                                        // reuse Xq as attn output

    // 1) q = q_w * high   (f32 in, bf16 out)
    gemm_wmma_k<true, false><<<dim3(HW / 128, CDIM / 128, BATCH), 256, 0, stream>>>(
        q_w, high, Xq, CDIM, HW, CDIM);
    // 2) kv = kv_w * low
    gemm_wmma_k<true, false><<<dim3(HW / 128, (2 * CDIM) / 128, BATCH), 256, 0, stream>>>(
        kv_w, low, Xkv, 2 * CDIM, HW, CDIM);
    // 3) depthwise 3x3
    dwconv3x3_k<<<dim3(HW / 256, CDIM, BATCH), 256, 0, stream>>>(Xq, q_dw_w, Qb, CDIM);
    dwconv3x3_k<<<dim3(HW / 256, 2 * CDIM, BATCH), 256, 0, stream>>>(Xkv, kv_dw_w, KVb, 2 * CDIM);
    // 4) norms
    rownorm_k<<<2 * BATCH * CDIM, 256, 0, stream>>>(Qb, KVb, RQ, RK);
    // 5) scores (split-K atomics, zero first)
    hipMemsetAsync(S, 0, (size_t)32 * CH * CH * 4, stream);
    attn_qk_k<<<dim3(25, BATCH * HEADS), 256, 0, stream>>>(Qb, KVb, S);
    // 6) softmax
    softmax_k<<<128, 256, 0, stream>>>(S, RQ, RK, temp, P);
    // 7) out = P * v
    attn_v_k<<<dim3(HW / 128, BATCH * HEADS), 256, 0, stream>>>(P, KVb, OA);
    // 8) proj (bf16 in, f32 out -> d_out)
    gemm_wmma_k<false, true><<<dim3(HW / 128, CDIM / 128, BATCH), 256, 0, stream>>>(
        proj_w, OA, out, CDIM, HW, CDIM);
}
